// RNN_gru_52810917872234
// MI455X (gfx1250) — matrix operand order repaired and checked by reading, not compiled, not run on hardware
//
#include <hip/hip_runtime.h>
#include <hip/hip_bf16.h>

// ---------------------------------------------------------------------------
// GRU on gfx1250 (MI455X): bf16 WMMA GEMMs (v_wmma_f32_16x16x32_bf16),
// async global->LDS weight staging (ASYNCcnt), LDS-resident B operands,
// persistent-scan kernel with grid barrier, packed B-fragment weight layout.
// ---------------------------------------------------------------------------

typedef __bf16 bf16;
typedef __attribute__((ext_vector_type(8)))  __bf16 v8bf;
typedef __attribute__((ext_vector_type(16))) __bf16 v16bf;
typedef __attribute__((ext_vector_type(8)))  float  v8f;

constexpr int T_STEPS = 512;
constexpr int BB      = 64;    // batch
constexpr int ID      = 1024;  // input dim
constexpr int HD      = 1024;  // hidden dim
constexpr int OD      = 1024;  // output dim
constexpr long MIN_   = (long)T_STEPS * BB;  // 32768 rows for the big GEMMs

// ---------------------------------------------------------------------------
// WMMA fragment helpers (wave32, 16x16x32 bf16, f32 accumulate)
//
// A (16x32 bf16): lane L holds row M = L&15; K-base = (L&16) ? 16 : 0.
//   elements j=0..7  -> K = kbase + j        (contiguous 16B)
//   elements j=8..15 -> K = kbase + j        (contiguous 16B)
// B (32x16 bf16): lane L holds col N = L&15; K-base = (L&16) ? 16 : 0,
//   elements j=0..15 -> K = kbase + j  (pre-packed contiguous by pack_w)
// C/D (16x16 f32): element g: row = g + ((L&16)?8:0), col = L&15.
// ---------------------------------------------------------------------------

__device__ __forceinline__ v16bf load_a_frag(const bf16* __restrict__ tile,
                                             int lda, int lane) {
  // tile points at (row 0, k 0) of the 16x32 sub-tile, row-major, stride lda
  const bf16* p = tile + (long)(lane & 15) * lda + ((lane & 16) ? 16 : 0);
  v8bf lo = *(const v8bf*)(p);
  v8bf hi = *(const v8bf*)(p + 8);
  v16bf a;
#pragma unroll
  for (int i = 0; i < 8; ++i) { a[i] = lo[i]; a[i + 8] = hi[i]; }
  return a;
}

// B fragment from an LDS panel (packed layout): frag = 32 lanes x 16 bf16
__device__ __forceinline__ v16bf lds_b_frag(const bf16* panel, int frag,
                                            int lane) {
  const bf16* q = panel + ((long)frag * 32 + lane) * 16;
  v8bf lo = *(const v8bf*)(q);
  v8bf hi = *(const v8bf*)(q + 8);
  v16bf b;
#pragma unroll
  for (int i = 0; i < 8; ++i) { b[i] = lo[i]; b[i + 8] = hi[i]; }
  return b;
}

__device__ __forceinline__ v8f wmma_bf16(v16bf a, v16bf b, v8f c) {
  return __builtin_amdgcn_wmma_f32_16x16x32_bf16(
      /*neg_a=*/false, a, /*neg_b=*/false, b,
      /*c_mod=*/(short)0, c, /*reuse_a=*/false, /*reuse_b=*/false);
}

__device__ __forceinline__ float sigmoidf_(float x) {
  return 1.0f / (1.0f + __expf(-x));
}

// ---------------------------------------------------------------------------
// Async global -> LDS copy (CDNA5 GLOBAL_LOAD_ASYNC_TO_LDS, ASYNCcnt).
// lds_off is the wave-relative LDS byte offset (low 32 bits of a flat LDS
// address, per the aperture mapping).
// ---------------------------------------------------------------------------
__device__ __forceinline__ void async_ld_lds_b128(unsigned lds_off,
                                                  const void* gptr) {
  asm volatile("global_load_async_to_lds_b128 %0, %1, off"
               :
               : "v"(lds_off), "v"(gptr)
               : "memory");
}

__device__ __forceinline__ void wait_async_zero() {
  asm volatile("s_wait_asynccnt 0x0" ::: "memory");
}

__device__ __forceinline__ unsigned lds_offset_of(const void* shared_ptr) {
  // flat addresses in the LDS aperture use addr[31:0] as the LDS address
  return (unsigned)(size_t)shared_ptr;
}

// ---------------------------------------------------------------------------
// Grid-wide sense-reversing barrier (co-resident workgroups)
// ---------------------------------------------------------------------------
__device__ __forceinline__ void grid_sync(unsigned* cnt, unsigned* gen,
                                          unsigned nwg) {
  __syncthreads();
  if (threadIdx.x == 0) {
    __threadfence();
    unsigned g = __hip_atomic_load(gen, __ATOMIC_ACQUIRE,
                                   __HIP_MEMORY_SCOPE_AGENT);
    unsigned a = __hip_atomic_fetch_add(cnt, 1u, __ATOMIC_ACQ_REL,
                                        __HIP_MEMORY_SCOPE_AGENT);
    if (a == nwg - 1u) {
      __hip_atomic_store(cnt, 0u, __ATOMIC_RELAXED, __HIP_MEMORY_SCOPE_AGENT);
      __hip_atomic_fetch_add(gen, 1u, __ATOMIC_ACQ_REL,
                             __HIP_MEMORY_SCOPE_AGENT);
    } else {
      while (__hip_atomic_load(gen, __ATOMIC_ACQUIRE,
                               __HIP_MEMORY_SCOPE_AGENT) == g) {
        __builtin_amdgcn_s_sleep(2);
      }
    }
    __threadfence();
  }
  __syncthreads();
}

// ---------------------------------------------------------------------------
// fp32 -> bf16 elementwise convert
// ---------------------------------------------------------------------------
__global__ void cvt_f32_bf16(const float* __restrict__ in,
                             bf16* __restrict__ out, long n) {
  for (long i = (long)blockIdx.x * blockDim.x + threadIdx.x; i < n;
       i += (long)gridDim.x * blockDim.x) {
    out[i] = (bf16)in[i];
  }
}

// ---------------------------------------------------------------------------
// Pack a row-major fp32 [K,N] weight into per-lane B-fragment bf16 layout:
// packed[((ct*KT + kt)*32 + lane)*16 + j] = W[kt*32 + ((lane&16)?16:0) + j,
//                                             ct*16 + (lane&15)]
// ---------------------------------------------------------------------------
__global__ void pack_w(const float* __restrict__ W, bf16* __restrict__ out,
                       int K, int N) {
  const long n  = (long)K * N;
  const int  KT = K >> 5;
  for (long idx = (long)blockIdx.x * blockDim.x + threadIdx.x; idx < n;
       idx += (long)gridDim.x * blockDim.x) {
    int  j    = (int)(idx & 15);
    int  lane = (int)((idx >> 4) & 31);
    long rest = idx >> 9;
    int  kt   = (int)(rest % KT);
    int  ct   = (int)(rest / KT);
    int  k    = kt * 32 + ((lane & 16) ? 16 : 0) + j;
    int  col  = ct * 16 + (lane & 15);
    out[idx]  = (bf16)W[(long)k * N + col];
  }
}

// ---------------------------------------------------------------------------
// Big GEMM: C[z] = A @ Wpk[z] + bias[z].  A: bf16 [M,K] row-major.
// Block: 256 threads (8 waves). WG tile: 128 rows x 64 cols.
// The WG's whole packed B panel (contiguous 128 KB) is async-copied into LDS
// once; the K-loop then streams A from global (b128) and B from LDS (b128).
// grid = (M/128, N/64, nz); Wpk z-stride = K*N, C z-stride = M*N.
// ---------------------------------------------------------------------------
__global__ void gemm_bf16_wmma(const bf16* __restrict__ A,
                               const bf16* __restrict__ Wpk,
                               const float* __restrict__ b0,
                               const float* __restrict__ b1,
                               const float* __restrict__ b2,
                               float* __restrict__ C, int M, int K, int N) {
  extern __shared__ bf16 smem[];  // 4 col-subtiles * KT frags * 1KB = 128 KB

  const int lane = threadIdx.x & 31;
  const int w    = threadIdx.x >> 5;
  const int z    = blockIdx.z;

  const bf16*  Wz   = Wpk + (long)z * K * N;
  const float* bias = (z == 0) ? b0 : (z == 1) ? b1 : b2;
  float*       Cz   = C + (long)z * M * N;

  const int r0  = blockIdx.x * 128 + w * 16;
  const int c0  = blockIdx.y * 64;
  const int ct0 = c0 >> 4;
  const int KT  = K >> 5;

  // ---- async preload of the packed B panel (contiguous in global) ----
  {
    const bf16* src    = Wz + (long)ct0 * KT * 512;  // 512 bf16 per fragment
    const int   chunks = (4 * KT * 512) / 8;         // b128 = 8 bf16
    for (int i = threadIdx.x; i < chunks; i += blockDim.x) {
      async_ld_lds_b128(lds_offset_of(smem + (long)i * 8), src + (long)i * 8);
    }
    wait_async_zero();
    __syncthreads();
  }

  v8f acc0 = {}, acc1 = {}, acc2 = {}, acc3 = {};
  const bf16* Arow = A + (long)r0 * K;

  for (int kt = 0; kt < KT; ++kt) {
    // WGP-scope prefetch of the A stream (pull into near caches)
    __builtin_prefetch(Arow + (kt + 8) * 32, 0, 3);  // global_prefetch_b8
    v16bf a   = load_a_frag(Arow + kt * 32, K, lane);
    v16bf bf0 = lds_b_frag(smem, 0 * KT + kt, lane);
    v16bf bf1 = lds_b_frag(smem, 1 * KT + kt, lane);
    v16bf bf2 = lds_b_frag(smem, 2 * KT + kt, lane);
    v16bf bf3 = lds_b_frag(smem, 3 * KT + kt, lane);
    acc0 = wmma_bf16(a, bf0, acc0);
    acc1 = wmma_bf16(a, bf1, acc1);
    acc2 = wmma_bf16(a, bf2, acc2);
    acc3 = wmma_bf16(a, bf3, acc3);
  }

  const int rbase = r0 + ((lane & 16) ? 8 : 0);
  const int cbase = c0 + (lane & 15);
#pragma unroll
  for (int g = 0; g < 8; ++g) {
    float* crow = Cz + (long)(rbase + g) * N;
    crow[cbase +  0] = acc0[g] + bias[cbase +  0];
    crow[cbase + 16] = acc1[g] + bias[cbase + 16];
    crow[cbase + 32] = acc2[g] + bias[cbase + 32];
    crow[cbase + 48] = acc3[g] + bias[cbase + 48];
  }
}

// ---------------------------------------------------------------------------
// Persistent GRU scan. 32 WGs x 256 threads; WG owns 32 state columns.
// The WG's recurrent-weight slice (3 gates x 64 KB = 192 KB) is async-copied
// into LDS once; all 512 steps then read B operands at LDS latency.
// Wave layout: rows = (w&3)*16, col subtile = (w>>2).
// Phase A: R,U = sigmoid(x + h@W);  hR = h*R  (bf16).   [grid barrier]
// Phase B: Hc = tanh(x + hR@Whh); h' = U*h + (1-U)*Hc.  [grid barrier]
// U stays in registers across the intra-step barrier.
// ---------------------------------------------------------------------------
__global__ void gru_scan(const bf16* __restrict__ WhrPk,
                         const bf16* __restrict__ WhuPk,
                         const bf16* __restrict__ WhhPk,
                         const float* __restrict__ Xr,
                         const float* __restrict__ Xu,
                         const float* __restrict__ Xh,
                         float* __restrict__ h32, bf16* __restrict__ hbf,
                         bf16* __restrict__ hR, bf16* __restrict__ Sbf,
                         float* __restrict__ states, unsigned* bar_cnt,
                         unsigned* bar_gen) {
  extern __shared__ bf16 smem[];  // 3 panels x 32768 bf16 (64 KB each)

  const int lane = threadIdx.x & 31;
  const int w    = threadIdx.x >> 5;
  const int c0   = blockIdx.x * 32;       // WG column base
  const int r0   = (w & 3) * 16;          // wave row base
  const int ctl  = w >> 2;                // local col-subtile (0 or 1)
  const int KT   = HD >> 5;               // 32 K-iterations
  const unsigned nwg = gridDim.x;
  const int PANEL = 2 * KT * 512;         // 32768 bf16 per gate panel

  // ---- async preload of the three gate-weight panels ----
  {
    const bf16* gsrc[3] = {WhrPk, WhuPk, WhhPk};
#pragma unroll
    for (int gte = 0; gte < 3; ++gte) {
      const bf16* src = gsrc[gte] + (long)(c0 >> 4) * KT * 512;
      bf16*       dst = smem + (long)gte * PANEL;
      for (int i = threadIdx.x; i < PANEL / 8; i += blockDim.x) {
        async_ld_lds_b128(lds_offset_of(dst + (long)i * 8),
                          src + (long)i * 8);
      }
    }
  }

  // zero-init owned columns of the state (overlaps with async copies)
  for (int i = threadIdx.x; i < 64 * 32; i += blockDim.x) {
    int r = i >> 5, c = c0 + (i & 31);
    h32[(long)r * HD + c] = 0.0f;
    hbf[(long)r * HD + c] = (bf16)0.0f;
  }
  wait_async_zero();
  __syncthreads();
  grid_sync(bar_cnt, bar_gen, nwg);

  const bf16* panR = smem + 0 * PANEL;
  const bf16* panU = smem + 1 * PANEL;
  const bf16* panH = smem + 2 * PANEL;
  const int   fb   = ctl * KT;  // fragment base for this wave's subtile

  const int rb = r0 + ((lane & 16) ? 8 : 0);
  const int cA = c0 + ctl * 16 + (lane & 15);

  for (int t = 0; t < T_STEPS; ++t) {
    const float* xr = Xr + (long)t * (BB * HD);
    const float* xu = Xu + (long)t * (BB * HD);
    const float* xh = Xh + (long)t * (BB * HD);

    // ---- Phase A: R and U gates ----
    v8f ar = {}, au = {};
    for (int kt = 0; kt < KT; ++kt) {
      v16bf a = load_a_frag(hbf + (long)r0 * HD + kt * 32, HD, lane);
      ar = wmma_bf16(a, lds_b_frag(panR, fb + kt, lane), ar);
      au = wmma_bf16(a, lds_b_frag(panU, fb + kt, lane), au);
    }

    v8f U = {};
#pragma unroll
    for (int g = 0; g < 8; ++g) {
      long  base = (long)(rb + g) * HD;
      float R    = sigmoidf_(ar[g] + xr[base + cA]);
      U[g]       = sigmoidf_(au[g] + xu[base + cA]);
      hR[base + cA] = (bf16)(h32[base + cA] * R);
    }
    grid_sync(bar_cnt, bar_gen, nwg);

    // ---- Phase B: candidate and state update ----
    v8f ah = {};
    for (int kt = 0; kt < KT; ++kt) {
      v16bf a = load_a_frag(hR + (long)r0 * HD + kt * 32, HD, lane);
      ah = wmma_bf16(a, lds_b_frag(panH, fb + kt, lane), ah);
    }

    float* st = states + (long)t * (BB * HD);
    bf16*  sb = Sbf + (long)t * (BB * HD);
#pragma unroll
    for (int g = 0; g < 8; ++g) {
      long  base = (long)(rb + g) * HD;
      float hc   = tanhf(ah[g] + xh[base + cA]);
      float hn   = U[g] * h32[base + cA] + (1.0f - U[g]) * hc;
      h32[base + cA] = hn;
      bf16 bn = (bf16)hn;
      hbf[base + cA] = bn;
      sb[base + cA]  = bn;
      st[base + cA]  = hn;
    }
    grid_sync(bar_cnt, bar_gen, nwg);
  }
}

// ---------------------------------------------------------------------------
// Host launcher
// ---------------------------------------------------------------------------
extern "C" void kernel_launch(void* const* d_in, const int* in_sizes, int n_in,
                              void* d_out, int out_size, void* d_ws,
                              size_t ws_size, hipStream_t stream) {
  const float* X    = (const float*)d_in[0];
  const float* W_xr = (const float*)d_in[1];
  const float* W_xu = (const float*)d_in[2];
  const float* W_xh = (const float*)d_in[3];
  const float* W_hr = (const float*)d_in[4];
  const float* W_hu = (const float*)d_in[5];
  const float* W_hh = (const float*)d_in[6];
  const float* b_r  = (const float*)d_in[7];
  const float* b_u  = (const float*)d_in[8];
  const float* b_h  = (const float*)d_in[9];
  const float* W_hq = (const float*)d_in[10];
  const float* b_q  = (const float*)d_in[11];

  float* outputs = (float*)d_out;              // [T,B,O]
  float* states  = (float*)d_out + MIN_ * OD;  // [T,B,H]

  // ---- workspace carve-out ----
  char*  ws  = (char*)d_ws;
  size_t off = 0;
  auto take = [&](size_t bytes) -> char* {
    char* p = ws + off;
    off += (bytes + 255) & ~(size_t)255;
    return p;
  };
  bf16*     Xbf     = (bf16*)take(MIN_ * ID * sizeof(bf16));        // 64 MB
  bf16*     Sbf     = (bf16*)take(MIN_ * HD * sizeof(bf16));        // 64 MB
  bf16*     WxPk    = (bf16*)take(3L * ID * HD * sizeof(bf16));     // 6 MB
  bf16*     WhPk    = (bf16*)take(3L * HD * HD * sizeof(bf16));     // 6 MB
  bf16*     WqPk    = (bf16*)take((long)HD * OD * sizeof(bf16));    // 2 MB
  float*    Xproj   = (float*)take(3L * MIN_ * HD * sizeof(float)); // 384 MB
  bf16*     hRbuf   = (bf16*)take((long)BB * HD * sizeof(bf16));
  float*    h32     = (float*)take((long)BB * HD * sizeof(float));
  bf16*     hbf     = (bf16*)take((long)BB * HD * sizeof(bf16));
  unsigned* barrier = (unsigned*)take(64);

  // allow >64KB dynamic LDS (320 KB per WGP on gfx1250)
  hipFuncSetAttribute(reinterpret_cast<const void*>(gemm_bf16_wmma),
                      hipFuncAttributeMaxDynamicSharedMemorySize, 131072);
  hipFuncSetAttribute(reinterpret_cast<const void*>(gru_scan),
                      hipFuncAttributeMaxDynamicSharedMemorySize, 196608);

  // reset grid-barrier state every launch (graph-capture safe)
  hipMemsetAsync(barrier, 0, 64, stream);

  // ---- one-time conversions / weight packing ----
  cvt_f32_bf16<<<8192, 256, 0, stream>>>(X, Xbf, MIN_ * ID);
  pack_w<<<1024, 256, 0, stream>>>(W_xr, WxPk + 0L * ID * HD, ID, HD);
  pack_w<<<1024, 256, 0, stream>>>(W_xu, WxPk + 1L * ID * HD, ID, HD);
  pack_w<<<1024, 256, 0, stream>>>(W_xh, WxPk + 2L * ID * HD, ID, HD);
  pack_w<<<1024, 256, 0, stream>>>(W_hr, WhPk + 0L * HD * HD, HD, HD);
  pack_w<<<1024, 256, 0, stream>>>(W_hu, WhPk + 1L * HD * HD, HD, HD);
  pack_w<<<1024, 256, 0, stream>>>(W_hh, WhPk + 2L * HD * HD, HD, HD);
  pack_w<<<1024, 256, 0, stream>>>(W_hq, WqPk, HD, OD);

  // ---- input projections: Xproj[z] = X @ W_x{r,u,h} + b_{r,u,h} ----
  gemm_bf16_wmma<<<dim3(MIN_ / 128, HD / 64, 3), 256, 131072, stream>>>(
      Xbf, WxPk, b_r, b_u, b_h, Xproj, (int)MIN_, ID, HD);

  // ---- recurrent scan (persistent, 32 WGs, weights LDS-resident) ----
  const float* Xr = Xproj + 0L * MIN_ * HD;
  const float* Xu = Xproj + 1L * MIN_ * HD;
  const float* Xh = Xproj + 2L * MIN_ * HD;
  gru_scan<<<HD / 32, 256, 196608, stream>>>(
      WhPk + 0L * HD * HD, WhPk + 1L * HD * HD, WhPk + 2L * HD * HD, Xr, Xu,
      Xh, h32, hbf, hRbuf, Sbf, states, barrier, barrier + 16);

  // ---- output projection: outputs = states @ W_hq + b_q ----
  gemm_bf16_wmma<<<dim3(MIN_ / 128, OD / 64, 1), 256, 131072, stream>>>(
      Sbf, WqPk, b_q, b_q, b_q, outputs, (int)MIN_, HD, OD);
}
